// MartingaleAwareAttention_73108933312697
// MI455X (gfx1250) — matrix operand order repaired and checked by reading, not compiled, not run on hardware
//
#include <hip/hip_runtime.h>
#include <hip/hip_bf16.h>

#define D_MODEL 1024
#define N_HEADS 16
#define HEAD_DIM 64
#define BATCH 4
#define SEQ 512
#define ROWS (BATCH * SEQ) /* 2048 */
#define QK_SCALE 0.125f    /* 64^-0.5 */

typedef _Float16 v16h __attribute__((ext_vector_type(16)));
typedef _Float16 v8h  __attribute__((ext_vector_type(8)));
typedef float    v8f  __attribute__((ext_vector_type(8)));

// ---- WMMA fragment loaders (CDNA5 ISA 7.12.2 layouts, wave32) -------------
// A (16x32 f16): lane m = lane&15; halves 0..7 -> K = base0..base0+7,
//                halves 8..15 -> K = base0+16..base0+23, base0 = (lane<16?0:8)
__device__ __forceinline__ v16h load_a_frag(const _Float16* A, int ld, int lane) {
  const int m  = lane & 15;
  const int kb = (lane >> 4) << 3; // 0 or 8
  const _Float16* p = A + (size_t)m * ld + kb;
  v8h lo = *(const v8h*)(p);
  v8h hi = *(const v8h*)(p + 16);
  return __builtin_shufflevector(lo, hi, 0, 1, 2, 3, 4, 5, 6, 7,
                                         8, 9, 10, 11, 12, 13, 14, 15);
}

// B (32x16 f16): lane n = lane&15; halves 0..7 -> K = kb..kb+7,
//                halves 8..15 -> K = kb+16..kb+23, kb = (lane<16?0:8)
__device__ __forceinline__ v16h load_b_frag(const _Float16* B, int ld, int lane) {
  const int n  = lane & 15;
  const int kb = (lane >> 4) << 3;
  v16h r;
#pragma unroll
  for (int s = 0; s < 8; ++s) {
    r[s]     = B[(size_t)(kb + s) * ld + n];
    r[s + 8] = B[(size_t)(kb + 16 + s) * ld + n];
  }
  return r;
}

__device__ __forceinline__ v8f wmma_f16(v16h a, v16h b, v8f c) {
  return __builtin_amdgcn_wmma_f32_16x16x32_f16(false, a, false, b,
                                                (short)0, c, false, false);
}

// ---- f32 -> f16 conversion ------------------------------------------------
__global__ void cvt_kernel(const float* __restrict__ src,
                           _Float16* __restrict__ dst, int n) {
  for (int i = blockIdx.x * blockDim.x + threadIdx.x; i < n;
       i += gridDim.x * blockDim.x)
    dst[i] = (_Float16)src[i];
}

// ---- QKV projection: X(2048x1024) @ W(1024x1024) + b ----------------------
// 32x64 wave tile: 2 A-frags share 4 B-frags -> 8 WMMAs per 4 b128 A loads + 64 strided 16-bit B loads.
// z=0 -> Q (scaled by QK_SCALE, layout [b,h,s,d])
// z=1 -> K^T (layout [b,h,d,s], packed 16B stores)
// z=2 -> V (layout [b,h,s,d])
__global__ __launch_bounds__(256) void qkv_kernel(
    const _Float16* __restrict__ xh, const _Float16* __restrict__ wh,
    const float* __restrict__ bq, const float* __restrict__ bk,
    const float* __restrict__ bv, _Float16* __restrict__ qh,
    _Float16* __restrict__ kth, _Float16* __restrict__ vh) {
  const int lane = threadIdx.x & 31;
  const int wave = threadIdx.x >> 5;
  const int m0 = blockIdx.y * 256 + wave * 32;
  const int n0 = blockIdx.x * 64;
  const int z  = blockIdx.z;
  const _Float16* W = wh + (size_t)z * (D_MODEL * D_MODEL);
  const float* bias = (z == 0) ? bq : (z == 1) ? bk : bv;

  v8f acc[2][4] = {};
  const _Float16* Arow = xh + (size_t)m0 * D_MODEL;
  for (int k0 = 0; k0 < D_MODEL; k0 += 32) {
    const v16h a0 = load_a_frag(Arow + k0, D_MODEL, lane);
    const v16h a1 = load_a_frag(Arow + (size_t)16 * D_MODEL + k0, D_MODEL, lane);
    const _Float16* Wk = W + (size_t)k0 * D_MODEL + n0;
#pragma unroll
    for (int t = 0; t < 4; ++t) {
      const v16h bfr = load_b_frag(Wk + t * 16, D_MODEL, lane);
      acc[0][t] = wmma_f16(a0, bfr, acc[0][t]);
      acc[1][t] = wmma_f16(a1, bfr, acc[1][t]);
    }
  }

  const int b = m0 >> 9; // 512 rows per batch; 256-row blocks never straddle b
  const int rb = (lane >> 4) << 3;
#pragma unroll
  for (int u = 0; u < 2; ++u) {
    const int s_base = (m0 & 511) + u * 16 + rb;
#pragma unroll
    for (int t = 0; t < 4; ++t) {
      const int n = n0 + t * 16 + (lane & 15);
      const int hh = n >> 6, d = n & 63;
      const float bb = bias[n];
      if (z == 1) { // K^T: C-frag rows are consecutive s -> one 16B store
        v8h pk;
#pragma unroll
        for (int j = 0; j < 8; ++j) pk[j] = (_Float16)(acc[u][t][j] + bb);
        *(v8h*)(kth +
                (((size_t)(b * N_HEADS + hh) * HEAD_DIM + d) * SEQ + s_base)) = pk;
      } else {
        _Float16* dst = (z == 0) ? qh : vh;
        const float sc = (z == 0) ? QK_SCALE : 1.0f;
#pragma unroll
        for (int j = 0; j < 8; ++j) {
          const int s = s_base + j;
          dst[((size_t)(b * N_HEADS + hh) * SEQ + s) * HEAD_DIM + d] =
              (_Float16)((acc[u][t][j] + bb) * sc);
        }
      }
    }
  }
}

// ---- Flash attention: one (b,h) per blockIdx.y, 16 queries per wave -------
__global__ __launch_bounds__(256) void attn_kernel(
    const _Float16* __restrict__ qh, const _Float16* __restrict__ kth,
    const _Float16* __restrict__ vh, _Float16* __restrict__ oh) {
  __shared__ __align__(32) _Float16 Pbuf[8][16 * 32]; // per-wave P tile
  const int lane = threadIdx.x & 31;
  const int wave = threadIdx.x >> 5;
  const int bh = blockIdx.y;
  const int b = bh >> 4, h = bh & 15;
  const int q0 = blockIdx.x * 128 + wave * 16;

  const _Float16* Q  = qh  + ((size_t)bh * SEQ + q0) * HEAD_DIM;
  const _Float16* KT = kth + (size_t)bh * HEAD_DIM * SEQ;
  const _Float16* V  = vh  + (size_t)bh * SEQ * HEAD_DIM;
  _Float16* P = &Pbuf[wave][0];

  const v16h aq0 = load_a_frag(Q, HEAD_DIM, lane);      // d = 0..31
  const v16h aq1 = load_a_frag(Q + 32, HEAD_DIM, lane); // d = 32..63

  v8f accO[4] = {};
  float mrow[8], lrow[8];
#pragma unroll
  for (int j = 0; j < 8; ++j) { mrow[j] = -1e30f; lrow[j] = 0.f; }

  const int rb = (lane >> 4) << 3; // C-frag row base (0 or 8)
  const int nc = lane & 15;        // C-frag column

  for (int kk = 0; kk < SEQ; kk += 32) {
    // scores 16q x 32k (B-frag: lane = key column of K^T, halves = d)
    v8f sc0 = {}, sc1 = {};
    sc0 = wmma_f16(aq0, load_b_frag(KT + kk, SEQ, lane), sc0);
    sc0 = wmma_f16(aq1, load_b_frag(KT + 32 * SEQ + kk, SEQ, lane), sc0);
    sc1 = wmma_f16(aq0, load_b_frag(KT + kk + 16, SEQ, lane), sc1);
    sc1 = wmma_f16(aq1, load_b_frag(KT + 32 * SEQ + kk + 16, SEQ, lane), sc1);

    // online softmax (rows live across 16-lane halves in wave32)
    float c[8];
    v8f p0, p1;
#pragma unroll
    for (int j = 0; j < 8; ++j) {
      float t = fmaxf(sc0[j], sc1[j]);
      t = fmaxf(t, __shfl_xor(t, 1));
      t = fmaxf(t, __shfl_xor(t, 2));
      t = fmaxf(t, __shfl_xor(t, 4));
      t = fmaxf(t, __shfl_xor(t, 8));
      const float mnew = fmaxf(mrow[j], t);
      c[j] = __expf(mrow[j] - mnew);
      mrow[j] = mnew;
      p0[j] = __expf(sc0[j] - mnew);
      p1[j] = __expf(sc1[j] - mnew);
      float rs = p0[j] + p1[j];
      rs += __shfl_xor(rs, 1);
      rs += __shfl_xor(rs, 2);
      rs += __shfl_xor(rs, 4);
      rs += __shfl_xor(rs, 8);
      lrow[j] = lrow[j] * c[j] + rs;
    }
#pragma unroll
    for (int t = 0; t < 4; ++t)
#pragma unroll
      for (int j = 0; j < 8; ++j) accO[t][j] *= c[j];

    // C-layout -> A-layout via per-wave LDS bounce
#pragma unroll
    for (int j = 0; j < 8; ++j) {
      P[(rb + j) * 32 + nc]      = (_Float16)p0[j];
      P[(rb + j) * 32 + nc + 16] = (_Float16)p1[j];
    }
    asm volatile("s_wait_dscnt 0x0" ::: "memory"); // wave-private tile
    const v16h pa = load_a_frag(P, 32, lane);

    // O += P(16x32) @ V(32x64); B-frag: lane = d column of V, halves = key
#pragma unroll
    for (int t = 0; t < 4; ++t)
      accO[t] = wmma_f16(pa, load_b_frag(V + (size_t)kk * HEAD_DIM + t * 16,
                                         HEAD_DIM, lane), accO[t]);
  }

  // normalize and store O as [b, s, h*64+d] (row-major A input for out-proj)
#pragma unroll
  for (int t = 0; t < 4; ++t) {
    const int d = t * 16 + nc;
#pragma unroll
    for (int j = 0; j < 8; ++j) {
      const float val = accO[t][j] / lrow[j];
      const size_t row = (size_t)b * SEQ + q0 + rb + j;
      oh[row * D_MODEL + h * HEAD_DIM + d] = (_Float16)val;
    }
  }
}

// ---- Output projection with folded martingale coefficient -----------------
__global__ __launch_bounds__(256) void proj_kernel(
    const _Float16* __restrict__ ohh, const _Float16* __restrict__ woh,
    const float* __restrict__ bo, const float* __restrict__ var_w,
    const float* __restrict__ len_w, float* __restrict__ out) {
  const int lane = threadIdx.x & 31;
  const int wave = threadIdx.x >> 5;
  const int m0 = blockIdx.y * 256 + wave * 32;
  const int n0 = blockIdx.x * 64;

  v8f acc[2][4] = {};
  const _Float16* Arow = ohh + (size_t)m0 * D_MODEL;
  for (int k0 = 0; k0 < D_MODEL; k0 += 32) {
    const v16h a0 = load_a_frag(Arow + k0, D_MODEL, lane);
    const v16h a1 = load_a_frag(Arow + (size_t)16 * D_MODEL + k0, D_MODEL, lane);
    const _Float16* Wk = woh + (size_t)k0 * D_MODEL + n0;
#pragma unroll
    for (int t = 0; t < 4; ++t) {
      const v16h bfr = load_b_frag(Wk + t * 16, D_MODEL, lane);
      acc[0][t] = wmma_f16(a0, bfr, acc[0][t]);
      acc[1][t] = wmma_f16(a1, bfr, acc[1][t]);
    }
  }

  // adaptive = clamp(len_w * log(512)/512, 0.01, 1); perm branch == standard
  const float a1c = len_w[0] * (6.2383246250395075f / 512.0f);
  const float ad = fminf(fmaxf(a1c, 0.01f), 1.0f);
  const float coeff = (1.0f - ad) + ad * var_w[0];

  const int rb = (lane >> 4) << 3;
  const int nc = lane & 15;
#pragma unroll
  for (int u = 0; u < 2; ++u) {
#pragma unroll
    for (int t = 0; t < 4; ++t) {
      const int n = n0 + t * 16 + nc;
      const float bb = bo[n];
#pragma unroll
      for (int j = 0; j < 8; ++j)
        out[(size_t)(m0 + u * 16 + rb + j) * D_MODEL + n] =
            coeff * (acc[u][t][j] + bb);
    }
  }
}

extern "C" void kernel_launch(void* const* d_in, const int* in_sizes, int n_in,
                              void* d_out, int out_size, void* d_ws,
                              size_t ws_size, hipStream_t stream) {
  (void)in_sizes; (void)n_in; (void)out_size; (void)ws_size;
  const float* x     = (const float*)d_in[0];
  const float* wq    = (const float*)d_in[1];
  const float* bq    = (const float*)d_in[2];
  const float* wk    = (const float*)d_in[3];
  const float* bk    = (const float*)d_in[4];
  const float* wv    = (const float*)d_in[5];
  const float* bv    = (const float*)d_in[6];
  const float* wo    = (const float*)d_in[7];
  const float* bo    = (const float*)d_in[8];
  const float* var_w = (const float*)d_in[9];
  const float* len_w = (const float*)d_in[10];
  // d_in[11] (perms) is dead: attention is permutation-equivariant, so the
  // permutation-averaged branch equals the standard branch exactly.

  char* ws = (char*)d_ws;
  _Float16* xh  = (_Float16*)(ws);                 // 4 MiB  [2048,1024]
  _Float16* wqh = (_Float16*)(ws + (4u << 20));    // 2 MiB each: wq|wk|wv
  _Float16* woh = (_Float16*)(ws + (10u << 20));   // 2 MiB
  _Float16* qh  = (_Float16*)(ws + (12u << 20));   // 4 MiB  [b,h,s,d] (scaled)
  _Float16* kth = (_Float16*)(ws + (16u << 20));   // 4 MiB  [b,h,d,s]
  _Float16* vh  = (_Float16*)(ws + (20u << 20));   // 4 MiB  [b,h,s,d]
  _Float16* ohh = (_Float16*)(ws + (24u << 20));   // 4 MiB  [b*s, 1024]

  cvt_kernel<<<256, 256, 0, stream>>>(x, xh, ROWS * D_MODEL);
  cvt_kernel<<<256, 256, 0, stream>>>(wq, wqh, D_MODEL * D_MODEL);
  cvt_kernel<<<256, 256, 0, stream>>>(wk, wqh + (1u << 20), D_MODEL * D_MODEL);
  cvt_kernel<<<256, 256, 0, stream>>>(wv, wqh + (2u << 20), D_MODEL * D_MODEL);
  cvt_kernel<<<256, 256, 0, stream>>>(wo, woh, D_MODEL * D_MODEL);

  qkv_kernel<<<dim3(16, 8, 3), 256, 0, stream>>>(xh, wqh, bq, bk, bv,
                                                 qh, kth, vh);
  attn_kernel<<<dim3(SEQ / 128, BATCH * N_HEADS), 256, 0, stream>>>(qh, kth,
                                                                    vh, ohh);
  proj_kernel<<<dim3(16, 8), 256, 0, stream>>>(ohh, woh, bo, var_w, len_w,
                                               (float*)d_out);
}
